// EfficientUpdateFormer_24670292148863
// MI455X (gfx1250) — matrix operand order repaired and checked by reading, not compiled, not run on hardware
//
#include <hip/hip_runtime.h>
#include <hip/hip_bf16.h>

#define HID  384
#define NHEAD 8
#define DH   48
#define TT   48
#define NPT  1024
#define NV   64
#define NTOT 1088

typedef __attribute__((ext_vector_type(16))) __bf16    v16bf;
typedef __attribute__((ext_vector_type(8)))  float     v8f;
typedef __attribute__((ext_vector_type(8)))  unsigned  v8u;

static __device__ __forceinline__ unsigned bfbits(float f) {   // bf16 RNE, low 16 bits
  unsigned u = __builtin_bit_cast(unsigned, f);
  u += 0x7fffu + ((u >> 16) & 1u);
  return u >> 16;
}
static __device__ __forceinline__ float bfval(unsigned bits16) {
  unsigned u = bits16 << 16;
  return __builtin_bit_cast(float, u);
}
// split x into hi (bf16 bits) and lo (bf16 bits of residual)
static __device__ __forceinline__ void split_u(float x, unsigned& h, unsigned& l) {
  h = bfbits(x);
  l = bfbits(x - bfval(h));
}
// pack two adjacent-K values: lo-K in bits[15:0], hi-K in bits[31:16]
static __device__ __forceinline__ void pack_pair(float v_lo, float v_hi,
                                                 unsigned& ph, unsigned& pl) {
  unsigned h0, l0, h1, l1;
  split_u(v_lo, h0, l0);
  split_u(v_hi, h1, l1);
  ph = (h1 << 16) | h0;
  pl = (l1 << 16) | l0;
}

// ---------------------------------------------------------------------------
// GEMM: C[M,N] = act(A[M,K] @ W[K,N] + bias) (+ residual R)
// bf16 WMMA, 2-term hi/lo split: 3 x v_wmma_f32_16x16x32_bf16 per K-step.
// Block = 256 threads = 8 waves; wave w -> rows [bx*128+w*16,+16) x 64 cols.
// B tile (32x64, shared by all 8 waves) staged once per block into LDS in
// K-pair-packed form matching the WMMA B-fragment register layout exactly:
//   BhP[kp][n] = bf16hi(K=2kp+1)<<16 | bf16hi(K=2kp)   (BlP likewise, lo term)
// so a fragment is 8 ds_load_b32 bit-cast straight to v16bf (no elem moves).
// ---------------------------------------------------------------------------
#define BST 68   // LDS row stride in u32 (mult of 4 -> aligned b128, no conflicts)
__global__ __launch_bounds__(256) void k_gemm(
    const float* __restrict__ A, const float* __restrict__ W,
    const float* __restrict__ bias, const float* __restrict__ R,
    float* __restrict__ Cp, int M, int N, int K, int act)
{
  __shared__ unsigned BhP[16 * BST];
  __shared__ unsigned BlP[16 * BST];
  const int tid  = (int)threadIdx.x;
  const int lane = tid & 31;
  const int wv   = tid >> 5;
  const int l15  = lane & 15;
  const int half = lane >> 4;
  const int row0 = (blockIdx.x * 8 + wv) * 16;
  const int col0 = blockIdx.y * 64;
  const bool rowok = row0 < M;
  const int arow = rowok ? (row0 + l15) : l15;   // keep loads in-bounds, mask stores
  const float* ap = A + (size_t)arow * K;

  const int kp = tid >> 4;          // staging: k-pair 0..15 (K rows 2kp, 2kp+1)
  const int nq = (tid & 15) * 4;    // staging: col offset 0,4,...,60

  v8f acc[4];
  #pragma unroll
  for (int j = 0; j < 4; ++j)
    #pragma unroll
    for (int r = 0; r < 8; ++r) acc[j][r] = 0.f;

  const int Kmain = K & ~31;

  for (int kb = 0; kb < Kmain; kb += 32) {
    // ---- stage B tile (32 x 64) -> LDS, K-pair packed hi & lo ----
    {
      const float* w0 = W + (size_t)(kb + 2 * kp) * N + (col0 + nq);
      float4 r0 = *(const float4*)w0;        // K = 2kp
      float4 r1 = *(const float4*)(w0 + N);  // K = 2kp+1
      unsigned ph0, pl0, ph1, pl1, ph2, pl2, ph3, pl3;
      pack_pair(r0.x, r1.x, ph0, pl0);
      pack_pair(r0.y, r1.y, ph1, pl1);
      pack_pair(r0.z, r1.z, ph2, pl2);
      pack_pair(r0.w, r1.w, ph3, pl3);
      unsigned* bh = &BhP[kp * BST + nq];
      unsigned* bl = &BlP[kp * BST + nq];
      bh[0] = ph0; bh[1] = ph1; bh[2] = ph2; bh[3] = ph3;
      bl[0] = pl0; bl[1] = pl1; bl[2] = pl2; bl[3] = pl3;
      if (kb + 32 < Kmain)
        __builtin_prefetch(W + (size_t)(kb + 32 + 2 * kp) * N + (col0 + nq), 0, 1);
    }
    // ---- A fragment 16x32 (branchless b64 loads, pair-packed in u32) ----
    // lane->M=l15 ; u32 idx i (elements 2i,2i+1): K=16*half+2i{,+1}
    v16bf ah, al;
    {
      const float* a0 = ap + kb + 16 * half;
      const float* a1 = a0 + 8;
      v8u ahu, alu;
      #pragma unroll
      for (int i = 0; i < 4; ++i) {
        float2 x0 = *(const float2*)(a0 + 2 * i);
        float2 x1 = *(const float2*)(a1 + 2 * i);
        unsigned ph, pl;
        pack_pair(x0.x, x0.y, ph, pl); ahu[i] = ph;     alu[i] = pl;
        pack_pair(x1.x, x1.y, ph, pl); ahu[4 + i] = ph; alu[4 + i] = pl;
      }
      ah = __builtin_bit_cast(v16bf, ahu);
      al = __builtin_bit_cast(v16bf, alu);
    }
    __syncthreads();
    #pragma unroll
    for (int j = 0; j < 4; ++j) {
      // B fragment 32x16: lane->N=j*16+l15 ; u32 idx i -> kpair 8*half+i
      v8u bhu, blu;
      #pragma unroll
      for (int i = 0; i < 8; ++i) {
        bhu[i] = BhP[(8 * half + i) * BST + j * 16 + l15];
        blu[i] = BlP[(8 * half + i) * BST + j * 16 + l15];
      }
      v16bf bh = __builtin_bit_cast(v16bf, bhu);
      v16bf bl = __builtin_bit_cast(v16bf, blu);
      acc[j] = __builtin_amdgcn_wmma_f32_16x16x32_bf16(false, ah, false, bh, (short)0, acc[j], false, false);
      acc[j] = __builtin_amdgcn_wmma_f32_16x16x32_bf16(false, ah, false, bl, (short)0, acc[j], false, false);
      acc[j] = __builtin_amdgcn_wmma_f32_16x16x32_bf16(false, al, false, bh, (short)0, acc[j], false, false);
    }
    __syncthreads();
  }

  if (Kmain < K) {                      // guarded tail (only input proj, K=1110)
    const int kb = Kmain;
    {
      const float* w0 = W + (size_t)(kb + 2 * kp) * N + (col0 + nq);
      float4 r0, r1;
      if (kb + 2 * kp < K)     r0 = *(const float4*)w0;
      else                     { r0.x = r0.y = r0.z = r0.w = 0.f; }
      if (kb + 2 * kp + 1 < K) r1 = *(const float4*)(w0 + N);
      else                     { r1.x = r1.y = r1.z = r1.w = 0.f; }
      unsigned ph0, pl0, ph1, pl1, ph2, pl2, ph3, pl3;
      pack_pair(r0.x, r1.x, ph0, pl0);
      pack_pair(r0.y, r1.y, ph1, pl1);
      pack_pair(r0.z, r1.z, ph2, pl2);
      pack_pair(r0.w, r1.w, ph3, pl3);
      unsigned* bh = &BhP[kp * BST + nq];
      unsigned* bl = &BlP[kp * BST + nq];
      bh[0] = ph0; bh[1] = ph1; bh[2] = ph2; bh[3] = ph3;
      bl[0] = pl0; bl[1] = pl1; bl[2] = pl2; bl[3] = pl3;
    }
    v16bf ah, al;
    {
      v8u ahu, alu;
      #pragma unroll
      for (int i = 0; i < 4; ++i) {
        int k0 = kb + 16 * half + 2 * i;
        int k1 = kb + 16 * half + 8 + 2 * i;
        float x0a = (k0     < K) ? ap[k0]     : 0.f;
        float x0b = (k0 + 1 < K) ? ap[k0 + 1] : 0.f;
        float x1a = (k1     < K) ? ap[k1]     : 0.f;
        float x1b = (k1 + 1 < K) ? ap[k1 + 1] : 0.f;
        unsigned ph, pl;
        pack_pair(x0a, x0b, ph, pl); ahu[i] = ph;     alu[i] = pl;
        pack_pair(x1a, x1b, ph, pl); ahu[4 + i] = ph; alu[4 + i] = pl;
      }
      ah = __builtin_bit_cast(v16bf, ahu);
      al = __builtin_bit_cast(v16bf, alu);
    }
    __syncthreads();
    #pragma unroll
    for (int j = 0; j < 4; ++j) {
      v8u bhu, blu;
      #pragma unroll
      for (int i = 0; i < 8; ++i) {
        bhu[i] = BhP[(8 * half + i) * BST + j * 16 + l15];
        blu[i] = BlP[(8 * half + i) * BST + j * 16 + l15];
      }
      v16bf bh = __builtin_bit_cast(v16bf, bhu);
      v16bf bl = __builtin_bit_cast(v16bf, blu);
      acc[j] = __builtin_amdgcn_wmma_f32_16x16x32_bf16(false, ah, false, bh, (short)0, acc[j], false, false);
      acc[j] = __builtin_amdgcn_wmma_f32_16x16x32_bf16(false, ah, false, bl, (short)0, acc[j], false, false);
      acc[j] = __builtin_amdgcn_wmma_f32_16x16x32_bf16(false, al, false, bh, (short)0, acc[j], false, false);
    }
  }

  if (!rowok) return;
  // C/D layout: VGPR r -> M = row0 + r + 8*half, N = col0 + j*16 + l15
  #pragma unroll
  for (int j = 0; j < 4; ++j) {
    int n = col0 + j * 16 + l15;
    float bv = bias ? bias[n] : 0.f;
    #pragma unroll
    for (int r = 0; r < 8; ++r) {
      int mm = row0 + r + 8 * half;
      size_t off = (size_t)mm * N + n;
      float v = acc[j][r] + bv;
      if (R) v += R[off];
      if (act) {           // tanh-approx GELU
        float x3 = v * v * v;
        v = 0.5f * v * (1.f + tanhf(0.7978845608028654f * (v + 0.044715f * x3)));
      }
      Cp[off] = v;
    }
  }
}

// ---------------------------------------------------------------------------
// LayerNorm over last dim 384 (optionally * g + b). One 128-thread block / row.
// ---------------------------------------------------------------------------
__global__ __launch_bounds__(128) void k_ln(
    const float* __restrict__ x, float* __restrict__ y,
    const float* __restrict__ g, const float* __restrict__ b)
{
  const int row = blockIdx.x, t = threadIdx.x;
  const float* xr = x + (size_t)row * HID;
  float v0 = xr[t], v1 = xr[t + 128], v2 = xr[t + 256];
  __shared__ float red[128];
  red[t] = v0 + v1 + v2; __syncthreads();
  #pragma unroll
  for (int o = 64; o > 0; o >>= 1) { if (t < o) red[t] += red[t + o]; __syncthreads(); }
  float mu = red[0] * (1.f / HID);
  __syncthreads();
  float d0 = v0 - mu, d1 = v1 - mu, d2 = v2 - mu;
  red[t] = d0 * d0 + d1 * d1 + d2 * d2; __syncthreads();
  #pragma unroll
  for (int o = 64; o > 0; o >>= 1) { if (t < o) red[t] += red[t + o]; __syncthreads(); }
  float rs = rsqrtf(red[0] * (1.f / HID) + 1e-6f);
  float* yr = y + (size_t)row * HID;
  if (g) {
    yr[t]       = d0 * rs * g[t]       + b[t];
    yr[t + 128] = d1 * rs * g[t + 128] + b[t + 128];
    yr[t + 256] = d2 * rs * g[t + 256] + b[t + 256];
  } else {
    yr[t] = d0 * rs; yr[t + 128] = d1 * rs; yr[t + 256] = d2 * rs;
  }
}

// ---------------------------------------------------------------------------
// Fused online-softmax attention. Q:[B*N1,384], KV:[B*N2,768] (k=0:384,v=384:768)
// grid = (ceil(N1/64), 8 heads, B); block = 64 threads; thread owns one query row.
// ---------------------------------------------------------------------------
#define KVS 52   // LDS row stride (multiple of 4 -> 16B-aligned b128 stores)
__global__ __launch_bounds__(64) void k_attn(
    const float* __restrict__ Q, const float* __restrict__ KV,
    float* __restrict__ O, int N1, int N2)
{
  const int b = blockIdx.z, h = blockIdx.y;
  const int i = blockIdx.x * 64 + (int)threadIdx.x;
  const bool active = i < N1;
  __shared__ float Ks[64 * KVS];
  __shared__ float Vs[64 * KVS];
  float qv[DH], acc[DH];
  if (active) {
    const float* qp = Q + ((size_t)b * N1 + i) * HID + h * DH;
    #pragma unroll
    for (int e = 0; e < DH; ++e) qv[e] = qp[e];
  }
  #pragma unroll
  for (int e = 0; e < DH; ++e) acc[e] = 0.f;
  float m = -3.0e38f, s = 0.f;
  const float scale = 0.14433756729740643f;   // 48^-0.5
  for (int jb = 0; jb < N2; jb += 64) {
    int jmax = N2 - jb; if (jmax > 64) jmax = 64;
    int t = (int)threadIdx.x;
    if (t < jmax) {
      const float* base = KV + ((size_t)b * N2 + jb + t) * (2 * HID) + h * DH;
      const float4* kp4 = (const float4*)base;
      const float4* vp4 = (const float4*)(base + HID);
      #pragma unroll
      for (int e = 0; e < DH / 4; ++e) {
        *(float4*)&Ks[t * KVS + 4 * e] = kp4[e];
        *(float4*)&Vs[t * KVS + 4 * e] = vp4[e];
      }
    }
    __syncthreads();
    if (active) {
      for (int jj = 0; jj < jmax; ++jj) {
        float xx = 0.f;
        #pragma unroll
        for (int e = 0; e < DH; ++e) xx += qv[e] * Ks[jj * KVS + e];
        xx *= scale;
        float mn = fmaxf(m, xx);
        float c0 = __expf(m - mn);
        float p  = __expf(xx - mn);
        s = s * c0 + p;
        #pragma unroll
        for (int e = 0; e < DH; ++e) acc[e] = acc[e] * c0 + p * Vs[jj * KVS + e];
        m = mn;
      }
    }
    __syncthreads();
  }
  if (active) {
    float inv = 1.f / s;
    float* op = O + ((size_t)b * N1 + i) * HID + h * DH;
    #pragma unroll
    for (int e = 0; e < DH; ++e) op[e] = acc[e] * inv;
  }
}

// ---------------------------------------------------------------------------
// Data-movement helpers
// ---------------------------------------------------------------------------
__global__ void k_perm(const float* __restrict__ in, float* __restrict__ out, int D0, int D1)
{ // out[d1][d0][c] = in[d0][d1][c]
  size_t idx = (size_t)blockIdx.x * 256 + threadIdx.x;
  size_t total = (size_t)D0 * D1 * HID;
  if (idx >= total) return;
  int c = (int)(idx % HID); size_t r = idx / HID;
  int d1 = (int)(r % D1);  int d0 = (int)(r / D1);
  out[((size_t)d1 * D0 + d0) * HID + c] = in[idx];
}
__global__ void k_split(const float* __restrict__ s, float* __restrict__ pts, float* __restrict__ vt)
{ // s:[T][1088][C] -> pts:[T][1024][C], vt:[T][64][C]
  size_t idx = (size_t)blockIdx.x * 256 + threadIdx.x;
  size_t total = (size_t)TT * NTOT * HID;
  if (idx >= total) return;
  int c = (int)(idx % HID); size_t r = idx / HID;
  int n = (int)(r % NTOT); int t = (int)(r / NTOT);
  float v = s[idx];
  if (n < NPT) pts[((size_t)t * NPT + n) * HID + c] = v;
  else         vt [((size_t)t * NV + (n - NPT)) * HID + c] = v;
}
__global__ void k_concat(const float* __restrict__ pts, const float* __restrict__ vt, float* __restrict__ s)
{
  size_t idx = (size_t)blockIdx.x * 256 + threadIdx.x;
  size_t total = (size_t)TT * NTOT * HID;
  if (idx >= total) return;
  int c = (int)(idx % HID); size_t r = idx / HID;
  int n = (int)(r % NTOT); int t = (int)(r / NTOT);
  s[idx] = (n < NPT) ? pts[((size_t)t * NPT + n) * HID + c]
                     : vt [((size_t)t * NV + (n - NPT)) * HID + c];
}
__global__ void k_fillvirt(const float* __restrict__ v, float* __restrict__ tokens)
{ // tokens rows [NPT*TT, NTOT*TT): row = vi*TT + t  <-  v[vi][c]
  size_t idx = (size_t)blockIdx.x * 256 + threadIdx.x;
  size_t total = (size_t)NV * TT * HID;
  if (idx >= total) return;
  int c = (int)(idx % HID); size_t r = idx / HID;
  int vi = (int)(r / TT);
  tokens[((size_t)NPT * TT + r) * HID + c] = v[(size_t)vi * HID + c];
}
// Output heads: out[row][0:2]=x@fw+fb, [2:4]=x@vw+vb. One wave per row.
__global__ __launch_bounds__(256) void k_heads(
    const float* __restrict__ x, const float* __restrict__ fw, const float* __restrict__ fb,
    const float* __restrict__ vw, const float* __restrict__ vb,
    float* __restrict__ out, int rows)
{
  int gw = (int)((blockIdx.x * 256 + threadIdx.x) >> 5);
  int lane = (int)(threadIdx.x & 31);
  if (gw >= rows) return;
  const float* xr = x + (size_t)gw * HID;
  float a0 = 0, a1 = 0, a2 = 0, a3 = 0;
  for (int e = lane; e < HID; e += 32) {
    float xv = xr[e];
    a0 += xv * fw[2 * e];  a1 += xv * fw[2 * e + 1];
    a2 += xv * vw[2 * e];  a3 += xv * vw[2 * e + 1];
  }
  #pragma unroll
  for (int o = 16; o > 0; o >>= 1) {
    a0 += __shfl_down(a0, o); a1 += __shfl_down(a1, o);
    a2 += __shfl_down(a2, o); a3 += __shfl_down(a3, o);
  }
  if (lane == 0) {
    float* op = out + (size_t)gw * 4;
    op[0] = a0 + fb[0]; op[1] = a1 + fb[1]; op[2] = a2 + vb[0]; op[3] = a3 + vb[1];
  }
}

// ---------------------------------------------------------------------------
// Host orchestration
// ---------------------------------------------------------------------------
struct BP { const float *wq,*bq,*wkv,*bkv,*wo,*bo,*w1,*b1,*w2,*b2,*cg,*cb; };

extern "C" void kernel_launch(void* const* d_in, const int* in_sizes, int n_in,
                              void* d_out, int out_size, void* d_ws, size_t ws_size,
                              hipStream_t stream)
{
  const float *x_in=nullptr,*virt=nullptr,*iw=nullptr,*ib=nullptr;
  const float *fw=nullptr,*fb=nullptr,*vw=nullptr,*vb=nullptr;
  BP tb[3], sv[3], v2p[3], p2v[3];
  auto F = [&](int i){ return (const float*)d_in[i]; };

  if (n_in >= 2 && in_sizes[1] == NV * HID) {
    // dict insertion order flattening
    x_in=F(0); virt=F(1); iw=F(2); ib=F(3); fw=F(4); fb=F(5); vw=F(6); vb=F(7);
    int idx = 8;
    auto rd = [&](BP& bp, bool cross){
      bp.wq=F(idx++); bp.bq=F(idx++); bp.wkv=F(idx++); bp.bkv=F(idx++);
      bp.wo=F(idx++); bp.bo=F(idx++); bp.w1=F(idx++); bp.b1=F(idx++);
      bp.w2=F(idx++); bp.b2=F(idx++);
      if (cross){ bp.cg=F(idx++); bp.cb=F(idx++); } else { bp.cg=nullptr; bp.cb=nullptr; }
    };
    for (int i=0;i<3;i++) rd(tb[i],false);
    for (int i=0;i<3;i++) rd(sv[i],false);
    for (int i=0;i<3;i++) rd(v2p[i],true);
    for (int i=0;i<3;i++) rd(p2v[i],true);
  } else {
    // jax tree_flatten sorted-key order
    x_in=F(0); fb=F(1); fw=F(2); ib=F(3); iw=F(4);
    int idx = 5;
    auto rd = [&](BP& bp, bool cross){
      bp.b1=F(idx++); bp.b2=F(idx++); bp.bkv=F(idx++); bp.bo=F(idx++); bp.bq=F(idx++);
      if (cross){ bp.cb=F(idx++); bp.cg=F(idx++); } else { bp.cb=nullptr; bp.cg=nullptr; }
      bp.w1=F(idx++); bp.w2=F(idx++); bp.wkv=F(idx++); bp.wo=F(idx++); bp.wq=F(idx++);
    };
    for (int i=0;i<3;i++) rd(p2v[i],true);
    for (int i=0;i<3;i++) rd(sv[i],false);
    for (int i=0;i<3;i++) rd(tb[i],false);
    for (int i=0;i<3;i++) rd(v2p[i],true);
    vb=F(idx++); vw=F(idx++); virt=F(idx++);
  }

  // Workspace layout (fp32)
  float* ws = (float*)d_ws;
  const size_t R_TOK  = (size_t)NTOT * TT;           // 52224 rows
  const size_t SZ_TOK = R_TOK * HID;
  const int    HCH    = 13056;                       // MLP hidden chunk rows
  float* tok  = ws; ws += SZ_TOK;
  float* ybuf = ws; ws += SZ_TOK;
  float* qbuf = ws; ws += SZ_TOK;                    // also holds ctx_n temporarily
  float* kv   = ws; ws += R_TOK * 2 * HID;
  float* att  = ws; ws += SZ_TOK;
  float* hb   = ws; ws += (size_t)HCH * 4 * HID;
  float* sb   = ws; ws += SZ_TOK;
  float* pts  = ws; ws += (size_t)TT * NPT * HID;
  float* vt   = ws; ws += (size_t)TT * NV * HID;
  size_t need = (size_t)(ws - (float*)d_ws) * sizeof(float);
  if (ws_size < need) return;

  auto gemm = [&](const float* A, const float* W, const float* bias, const float* Rr,
                  float* Cc, int M, int N, int K, int act){
    dim3 g((unsigned)((M + 127) / 128), (unsigned)(N / 64), 1);
    k_gemm<<<g, dim3(256), 0, stream>>>(A, W, bias, Rr, Cc, M, N, K, act);
  };
  auto ln = [&](const float* xp, float* yp, int rows, const float* g, const float* b){
    k_ln<<<dim3((unsigned)rows), dim3(128), 0, stream>>>(xp, yp, g, b);
  };
  auto attn = [&](const float* Qp, const float* KVp, float* Op, int B_, int N1, int N2){
    dim3 g((unsigned)((N1 + 63) / 64), NHEAD, (unsigned)B_);
    k_attn<<<g, dim3(64), 0, stream>>>(Qp, KVp, Op, N1, N2);
  };
  auto mlp = [&](float* xp, int M, const BP& p){
    ln(xp, ybuf, M, nullptr, nullptr);
    for (int r0 = 0; r0 < M; r0 += HCH) {
      int mr = M - r0; if (mr > HCH) mr = HCH;
      gemm(ybuf + (size_t)r0 * HID, p.w1, p.b1, nullptr, hb, mr, 4 * HID, HID, 1);
      gemm(hb, p.w2, p.b2, xp + (size_t)r0 * HID, xp + (size_t)r0 * HID, mr, HID, 4 * HID, 0);
    }
  };
  auto attn_block = [&](float* xp, int B_, int S, const BP& p){
    int M = B_ * S;
    ln(xp, ybuf, M, nullptr, nullptr);
    gemm(ybuf, p.wq, p.bq, nullptr, qbuf, M, HID, HID, 0);
    gemm(ybuf, p.wkv, p.bkv, nullptr, kv, M, 2 * HID, HID, 0);
    attn(qbuf, kv, att, B_, S, S);
    gemm(att, p.wo, p.bo, xp, xp, M, HID, HID, 0);
    mlp(xp, M, p);
  };
  auto cross_block = [&](float* xp, const float* ctx, int B_, int N1, int N2, const BP& p){
    int M1 = B_ * N1, M2 = B_ * N2;
    ln(ctx, qbuf, M2, p.cg, p.cb);                       // ctx_n -> qbuf (temp)
    gemm(qbuf, p.wkv, p.bkv, nullptr, kv, M2, 2 * HID, HID, 0);
    ln(xp, ybuf, M1, nullptr, nullptr);
    gemm(ybuf, p.wq, p.bq, nullptr, qbuf, M1, HID, HID, 0);  // overwrites ctx_n (done)
    attn(qbuf, kv, att, B_, N1, N2);
    gemm(att, p.wo, p.bo, xp, xp, M1, HID, HID, 0);
    mlp(xp, M1, p);
  };

  const int MP = NPT * TT;                         // 49152 point-token rows
  // 1) input projection directly into tokens[:1024] ([n][t][c] layout)
  gemm(x_in, iw, ib, nullptr, tok, MP, HID, 1110, 0);
  // 2) broadcast virtual tracks into tokens[1024:]
  {
    size_t total = (size_t)NV * TT * HID;
    k_fillvirt<<<dim3((unsigned)((total + 255) / 256)), dim3(256), 0, stream>>>(virt, tok);
  }

  size_t totNT = R_TOK * HID;
  unsigned gNT = (unsigned)((totNT + 255) / 256);
  for (int d = 0; d < 3; ++d) {
    attn_block(tok, NTOT, TT, tb[d]);                       // time attention
    k_perm<<<gNT, 256, 0, stream>>>(tok, sb, NTOT, TT);     // [n][t] -> [t][n]
    k_split<<<gNT, 256, 0, stream>>>(sb, pts, vt);
    cross_block(vt, pts, TT, NV, NPT, v2p[d]);              // virtual <- points
    attn_block(vt, TT, NV, sv[d]);                          // virtual self
    cross_block(pts, vt, TT, NPT, NV, p2v[d]);              // points <- virtual
    k_concat<<<gNT, 256, 0, stream>>>(pts, vt, sb);
    k_perm<<<gNT, 256, 0, stream>>>(sb, tok, TT, NTOT);     // [t][n] -> [n][t]
  }

  // 3) flow/vis heads on point tokens, concatenated output (B,N,T,4)
  k_heads<<<dim3((unsigned)(((size_t)MP * 32 + 255) / 256)), dim3(256), 0, stream>>>(
      tok, fw, fb, vw, vb, (float*)d_out, MP);
  (void)out_size; (void)n_in;
}
